// ScaledLinearAttention_23081154248845
// MI455X (gfx1250) — matrix operand order repaired and checked by reading, not compiled, not run on hardware
//
#include <hip/hip_runtime.h>
#include <hip/hip_bf16.h>

// ScaledLinearAttention for MI455X (gfx1250).
//
// Algebraic collapse (qk head-dim == 1):
//   out[c,f] = bo[f] + sum_h qn[h,c] * M'[h,f]
//   M'[h,f]  = qscale[h] * sum_d ktv[h,d] * Wo[f, h*32+d]
//   ktv[h,d] = kscale[h] * sum_e y[h,e] * Wv[h*32+d, e] + bv[h*32+d]
//   y[h,e]   = sum_c k_elu[h,c] * x[c,e]        (normalization hoisted; sum_c kn == 1)
// => total ~3.3 GFLOP vs ~400 MB traffic: memory-bound (~17us at 23.3 TB/s).
//
// WMMA (v_wmma_f32_16x16x32_f16) computes QK = [Wq;Wk] @ X^T per 16-col tile.

#define BB 2
#define RR 190
#define CC 512
#define EE 256
#define HH 8
#define DD 32

typedef __attribute__((ext_vector_type(16))) _Float16 v16h;
typedef __attribute__((ext_vector_type(8)))  _Float16 v8h;
typedef __attribute__((ext_vector_type(8)))  float    v8f;
typedef __attribute__((ext_vector_type(4)))  float    f32x4;

// LDS pitches chosen for bank-conflict avoidance (64 banks x 4B).
#define WQK_PITCH 264   // f16 elems per row of staged [Wq;Wk]  (528B rows, 16B aligned)
#define QK_PITCH  516   // f32 elems per row of elu'd q/k       (m*516 % 64 = m*4 -> distinct banks)

__global__ __launch_bounds__(512, 1)
void sla_kernel(const float* __restrict__ x,
                const float* __restrict__ Wq, const float* __restrict__ bq,
                const float* __restrict__ Wk, const float* __restrict__ bk,
                const float* __restrict__ Wv, const float* __restrict__ bv,
                const float* __restrict__ Wo, const float* __restrict__ bo,
                float* __restrict__ out)
{
    __shared__ _Float16 s_wqk[16 * WQK_PITCH];  // [Wq rows 0-7; Wk rows 8-15] as f16
    __shared__ float    s_qk [16 * QK_PITCH];   // elu(q)+1 rows 0-7, elu(k)+1 rows 8-15 (raw, unnormalized)
    __shared__ float    s_y  [HH * EE];         // y_raw[h][e]
    __shared__ float    s_ktv[HH * (DD + 1)];   // ktv[h][d] (padded)
    __shared__ float    s_mt [EE * 8];          // M'[f][h]  (transposed, qscale folded in)
    __shared__ float    s_sum[16];              // per-head sums of elu'd q/k over c
    __shared__ float    s_scale[16];            // [0..7]=C/sum_q, [8..15]=1/sum_k
    __shared__ float    s_bqk[16];

    const int tid  = threadIdx.x;
    const int lane = tid & 31;
    const int wave = tid >> 5;
    const int half = lane >> 4;   // which 16-lane half of the wave
    const int l16  = lane & 15;

    const long slice = (long)blockIdx.x * (long)(CC * EE);
    const float* X = x + slice;

    // ---- Stage [Wq;Wk] as f16 into LDS; init sums/biases ----
    for (int i = tid; i < 16 * EE; i += 512) {
        const int m = i >> 8;       // row 0..15
        const int e = i & 255;
        const float w = (m < 8) ? Wq[m * EE + e] : Wk[(m - 8) * EE + e];
        s_wqk[m * WQK_PITCH + e] = (_Float16)w;
    }
    if (tid < 16) {
        s_bqk[tid] = (tid < 8) ? bq[tid] : bk[tid - 8];
        s_sum[tid] = 0.0f;
    }
    __syncthreads();

    // ---- Pass 1: QK tile = Wqk @ X^T via WMMA; elu+1; head sums ----
    // A layout (16-bit 16x32): lane holds row m=l16; K chunk [16*half..+15] (matches B).
    // B layout (16-bit 32x16): lane holds col n=l16; K chunk [16*half..+15] -> 16 consecutive
    //   f32 of x row (cbase+l16): one full 64B cacheline per lane per K-step.
    for (int tile = wave; tile < CC / 16; tile += 16) {
        const int cbase = tile * 16;
        const float* xrow = X + (long)(cbase + l16) * EE;
        if (tile < 16) {
            __builtin_prefetch(X + (long)(cbase + 256 + l16) * EE, 0, 1); // next c-tile
        }
        const _Float16* arow = s_wqk + l16 * WQK_PITCH;

        v8f acc = {};
        for (int ek = 0; ek < EE / 32; ++ek) {
            const int ebase = ek * 32;
            // A operand from LDS
            const v8h a0 = *(const v8h*)(arow + ebase + 16 * half);
            const v8h a1 = *(const v8h*)(arow + ebase + 16 * half + 8);
            v16h a;
            #pragma unroll
            for (int i = 0; i < 8; ++i) { a[i] = a0[i]; a[8 + i] = a1[i]; }
            // B operand: 16 consecutive f32 from global, convert to f16
            const f32x4* bp = (const f32x4*)(xrow + ebase + 16 * half);
            const f32x4 b0 = bp[0], b1 = bp[1], b2 = bp[2], b3 = bp[3];
            v16h b;
            b[0]=(_Float16)b0.x;  b[1]=(_Float16)b0.y;  b[2]=(_Float16)b0.z;  b[3]=(_Float16)b0.w;
            b[4]=(_Float16)b1.x;  b[5]=(_Float16)b1.y;  b[6]=(_Float16)b1.z;  b[7]=(_Float16)b1.w;
            b[8]=(_Float16)b2.x;  b[9]=(_Float16)b2.y;  b[10]=(_Float16)b2.z; b[11]=(_Float16)b2.w;
            b[12]=(_Float16)b3.x; b[13]=(_Float16)b3.y; b[14]=(_Float16)b3.z; b[15]=(_Float16)b3.w;
            acc = __builtin_amdgcn_wmma_f32_16x16x32_f16(
                false, a, false, b, (short)0, acc, false, false);
        }
        // D layout: VGPR v, lane -> (m = v + 8*half, c = cbase + l16)
        #pragma unroll
        for (int v = 0; v < 8; ++v) {
            const int m = v + 8 * half;
            const float t = acc[v] + s_bqk[m];
            const float e1 = (t > 0.0f) ? (t + 1.0f) : __expf(t);   // elu(t)+1
            s_qk[m * QK_PITCH + cbase + l16] = e1;
            float r = e1;                       // reduce within 16-lane half
            r += __shfl_xor(r, 1);
            r += __shfl_xor(r, 2);
            r += __shfl_xor(r, 4);
            r += __shfl_xor(r, 8);
            if (l16 == 0) atomicAdd(&s_sum[m], r);
        }
    }
    __syncthreads();

    if (tid < 16) {
        const float s = s_sum[tid];
        s_scale[tid] = (tid < 8) ? ((float)CC / s) : (1.0f / s);
    }
    // (s_scale consumed only after later barriers)

    // ---- Pass 2: y_raw[h][e] = sum_c k_elu[h][c] * X[c][e]  (coalesced x re-read, L2-hot) ----
    {
        const int e     = tid & 255;
        const int chalf = tid >> 8;
        float acc[8] = {0.f,0.f,0.f,0.f,0.f,0.f,0.f,0.f};
        const float* kp = s_qk + 8 * QK_PITCH;   // raw elu'd k rows
        const float* xc = X + (long)(chalf * 256) * EE + e;
        for (int i = 0; i < 256; ++i) {
            const float xv = xc[(long)i * EE];
            const int   c  = chalf * 256 + i;
            #pragma unroll
            for (int h = 0; h < 8; ++h)
                acc[h] = fmaf(kp[h * QK_PITCH + c], xv, acc[h]);
        }
        if (chalf == 0) {
            #pragma unroll
            for (int h = 0; h < 8; ++h) s_y[h * EE + e] = acc[h];
        }
        __syncthreads();
        if (chalf == 1) {
            #pragma unroll
            for (int h = 0; h < 8; ++h) s_y[h * EE + e] += acc[h];
        }
        __syncthreads();
    }

    // ---- Pass 3a: ktv[h][d] = kscale[h]*sum_e y[h][e]*Wv[h*32+d][e] + bv ----
    if (tid < 256) {
        const int h = tid >> 5, d = tid & 31;
        const float* wvr = Wv + (long)(h * DD + d) * EE;
        const float* yr  = s_y + h * EE;
        float s = 0.0f;
        for (int e = 0; e < EE; e += 4) {
            const f32x4 w = *(const f32x4*)(wvr + e);
            s += yr[e] * w.x + yr[e+1] * w.y + yr[e+2] * w.z + yr[e+3] * w.w;
        }
        s_ktv[h * (DD + 1) + d] = s * s_scale[8 + h] + bv[h * DD + d];
    }
    __syncthreads();

    // ---- Pass 3b: M'[f][h] = qscale[h] * sum_d ktv[h][d]*Wo[f][h*32+d] ----
    {
        const int f = tid & 255;
        const int g = tid >> 8;                 // handles h = g*4 .. g*4+3
        const float* wor = Wo + (long)f * EE;
        #pragma unroll
        for (int j = 0; j < 4; ++j) {
            const int h = g * 4 + j;
            const float* kt = s_ktv + h * (DD + 1);
            const float* wo = wor + h * DD;
            float s = 0.0f;
            for (int d = 0; d < DD; d += 4) {
                const f32x4 w = *(const f32x4*)(wo + d);
                s += kt[d] * w.x + kt[d+1] * w.y + kt[d+2] * w.z + kt[d+3] * w.w;
            }
            s_mt[f * 8 + h] = s * s_scale[h];
        }
    }
    __syncthreads();

    // ---- Pass 4: out[c][f] = bo[f] + sum_h q_elu[h][c] * M'[f][h]  (coalesced stores) ----
    {
        const int f     = tid & 255;
        const int chalf = tid >> 8;
        float m[8];
        #pragma unroll
        for (int h = 0; h < 8; ++h) m[h] = s_mt[f * 8 + h];
        const float bof = bo[f];
        float* orow = out + slice + (long)(chalf * 256) * EE + f;
        for (int i = 0; i < 256; ++i) {
            const int c = chalf * 256 + i;
            float s = bof;
            #pragma unroll
            for (int h = 0; h < 8; ++h)
                s = fmaf(s_qk[h * QK_PITCH + c], m[h], s);
            orow[(long)i * EE] = s;
        }
    }
}

extern "C" void kernel_launch(void* const* d_in, const int* in_sizes, int n_in,
                              void* d_out, int out_size, void* d_ws, size_t ws_size,
                              hipStream_t stream) {
    const float* x  = (const float*)d_in[0];
    const float* Wq = (const float*)d_in[1];
    const float* bq = (const float*)d_in[2];
    const float* Wk = (const float*)d_in[3];
    const float* bk = (const float*)d_in[4];
    const float* Wv = (const float*)d_in[5];
    const float* bv = (const float*)d_in[6];
    const float* Wo = (const float*)d_in[7];
    const float* bo = (const float*)d_in[8];
    float* out = (float*)d_out;

    dim3 grid(BB * RR), block(512);
    sla_kernel<<<grid, block, 0, stream>>>(x, Wq, bq, Wk, bk, Wv, bv, Wo, bo, out);
}
